// MultiLatentAttention_51565377356070
// MI455X (gfx1250) — compile-verified, matrix operand order checked by reading, not run on hardware
//
#include <hip/hip_runtime.h>

// ---------------------------------------------------------------------------
// MLA (Multi-Latent Attention) for gfx1250 — bf16 WMMA + TDM async staging
// B=2, S=2048, D=2048, H=16, HD=128, RD=64, L=512
// ---------------------------------------------------------------------------

typedef __attribute__((ext_vector_type(16))) __bf16 v16bf;
typedef __attribute__((ext_vector_type(8)))  float  v8f;
typedef __attribute__((ext_vector_type(4)))  unsigned int v4u;
typedef __attribute__((ext_vector_type(8)))  int    v8i;
typedef __attribute__((ext_vector_type(4)))  int    v4i;

union FragU { v16bf v; uint4 q[2]; };

// Load a 16x32 bf16 WMMA A/B fragment from a row-major matrix.
// 16-bit A-layout (ISA 7.12.2): lanes 0-15 -> rows, K chunks {0..7,16..23};
// lanes 16-31 -> same rows, K chunks {8..15,24..31}. Two 16-byte loads/lane.
__device__ inline v16bf ldfrag(const __bf16* __restrict__ base, int row0, int ld, int k0) {
  const int lane = threadIdx.x & 31;
  const int half = lane >> 4;
  const __bf16* p = base + (size_t)(row0 + (lane & 15)) * (size_t)ld + k0 + 8 * half;
  FragU f;
  f.q[0] = *(const uint4*)(p);
  f.q[1] = *(const uint4*)(p + 16);
  return f.v;
}

__device__ inline v8f wmma_bf16(v16bf a, v16bf b, v8f c) {
  return __builtin_amdgcn_wmma_f32_16x16x32_bf16(false, a, false, b, (short)0, c,
                                                 false, false);
}

// ---------------------------------------------------------------------------
// TDM: 2D tensor_load_to_lds. D# packing per CDNA5 ISA.
//   group0: [1:0]=count=1, [63:32]=lds_addr, [120:64]=global_addr, [127:126]=2
//   group1: [17:16]=data_size(1=2B), tensor_dim0/1, tile_dim0/1, dim0_stride
// LDS aperture low 32 bits == LDS byte offset (ISA 10.2), so truncate the
// flat shared pointer for lds_addr. 6-arg builtin form (clang-23 toolchain).
// ---------------------------------------------------------------------------
__device__ inline void tdm_load_2d(void* lds_ptr, const void* gptr,
                                   unsigned tile_d0, unsigned tile_d1,
                                   unsigned long long stride_elems) {
  unsigned long long ga = (unsigned long long)gptr;
  const unsigned td0 = 0x100000u, td1 = 0x100000u;   // large tensor dims (in-bounds)
  v4u g0;
  g0[0] = 1u;                                        // count=1
  g0[1] = (unsigned)(size_t)lds_ptr;                 // lds_addr (bytes)
  g0[2] = (unsigned)(ga & 0xffffffffu);
  g0[3] = (unsigned)((ga >> 32) & 0x1ffffffu) | (2u << 30);  // type=2 (image)
  v8i g1;
  g1[0] = (int)(1u << 16);                           // data_size = 2 bytes
  g1[1] = (int)((td0 & 0xffffu) << 16);              // tensor_dim0[15:0]
  g1[2] = (int)((td0 >> 16) | ((td1 & 0xffffu) << 16));
  g1[3] = (int)((td1 >> 16) | (tile_d0 << 16));      // tile_dim0
  g1[4] = (int)tile_d1;                              // tile_dim1 (tile_dim2 = 0)
  g1[5] = (int)(unsigned)(stride_elems & 0xffffffffu);          // dim0 stride
  g1[6] = (int)(unsigned)((stride_elems >> 32) & 0xffffu);
  g1[7] = 0;
  v4i z4 = {0, 0, 0, 0};
  v8i z8 = {0, 0, 0, 0, 0, 0, 0, 0};
  __builtin_amdgcn_tensor_load_to_lds(g0, g1, z4, z4, z8, 0);
}

// ---------------------------------------------------------------------------
// fp32 -> bf16 elementwise convert (8 elements / thread, vectorized)
// ---------------------------------------------------------------------------
__global__ __launch_bounds__(256) void f2bf(const float* __restrict__ in,
                                            __bf16* __restrict__ out, int n) {
  int i = (blockIdx.x * 256 + threadIdx.x) * 8;
  if (i < n) {
    float4 a = *(const float4*)(in + i);
    float4 b = *(const float4*)(in + i + 4);
    __bf16 r[8] = {(__bf16)a.x, (__bf16)a.y, (__bf16)a.z, (__bf16)a.w,
                   (__bf16)b.x, (__bf16)b.y, (__bf16)b.z, (__bf16)b.w};
    *(uint4*)(out + i) = *(const uint4*)r;
  }
}

// ---------------------------------------------------------------------------
// fp32 -> bf16 transpose-convert: in K x N (row-major) -> out N x K.
// 32x32 LDS tile, coalesced loads and stores. K,N multiples of 32.
// ---------------------------------------------------------------------------
__global__ __launch_bounds__(256) void f2bfT(const float* __restrict__ in,
                                             __bf16* __restrict__ out, int K, int N) {
  __shared__ float tile[32][33];
  const int bx = blockIdx.x * 32;          // n
  const int by = blockIdx.y * 32;          // k
  const int tx = threadIdx.x & 31, ty = threadIdx.x >> 5;
#pragma unroll
  for (int i = 0; i < 4; ++i)
    tile[ty + i * 8][tx] = in[(size_t)(by + ty + i * 8) * N + bx + tx];
  __syncthreads();
#pragma unroll
  for (int i = 0; i < 4; ++i)
    out[(size_t)(bx + ty + i * 8) * K + by + tx] = (__bf16)tile[tx][ty + i * 8];
}

// ---------------------------------------------------------------------------
// Generic bf16 GEMM: C[M,N] = A[M,K] * BT[N,K]^T, fp32 accumulate.
// B is pre-transposed, so both operands stage as coalesced uint4 row copies.
// 256 threads = 8 waves (2x4 wave grid), 128x128 C tile, K-step 32.
// Ping-pong LDS (one barrier / K-step), register prefetch of next tile.
// ---------------------------------------------------------------------------
template <typename OutT>
__global__ __launch_bounds__(256) void gemm_bf16_128x128(
    const __bf16* __restrict__ A, const __bf16* __restrict__ BT,
    OutT* __restrict__ C, int N, int K) {
  __shared__ __bf16 sA[2][128 * 32];    // A tile,  row-major (m x K)
  __shared__ __bf16 sB[2][128 * 32];    // BT tile, row-major (n x K)

  const int tid  = threadIdx.x;
  const int wave = tid >> 5;
  const int lane = tid & 31;
  const int half = lane >> 4, lc = lane & 15;
  const int wm = wave >> 2, wn = wave & 3;   // 2 (M) x 4 (N) waves
  const int m0 = blockIdx.y * 128, n0 = blockIdx.x * 128;

  uint4 ra[2], rb[2];

  auto load_regs = [&](int k0) {
#pragma unroll
    for (int i = 0; i < 2; ++i) {
      int q = tid + 256 * i;
      int row = q >> 2, c4 = q & 3;
      ra[i] = *(const uint4*)(A  + (size_t)(m0 + row) * K + k0 + c4 * 8);
      rb[i] = *(const uint4*)(BT + (size_t)(n0 + row) * K + k0 + c4 * 8);
    }
  };
  auto store_lds = [&](int buf) {
#pragma unroll
    for (int i = 0; i < 2; ++i) {
      ((uint4*)sA[buf])[tid + 256 * i] = ra[i];
      ((uint4*)sB[buf])[tid + 256 * i] = rb[i];
    }
  };

  v8f acc[4][2] = {};

  load_regs(0);
  store_lds(0);
  __syncthreads();

  const int nk = K >> 5;
  for (int kt = 0; kt < nk; ++kt) {
    const int buf = kt & 1;
    if (kt + 1 < nk) load_regs((kt + 1) << 5);   // overlap VMEM with WMMA

    v16bf bf0 = ldfrag(sB[buf], wn * 32 + 0,  32, 0);
    v16bf bf1 = ldfrag(sB[buf], wn * 32 + 16, 32, 0);
#pragma unroll
    for (int mi = 0; mi < 4; ++mi) {
      v16bf af = ldfrag(sA[buf], wm * 64 + mi * 16, 32, 0);
      acc[mi][0] = wmma_bf16(af, bf0, acc[mi][0]);
      acc[mi][1] = wmma_bf16(af, bf1, acc[mi][1]);
    }

    if (kt + 1 < nk) store_lds(buf ^ 1);         // disjoint buffer
    __syncthreads();                             // publish buf^1
  }

  // Epilogue. C layout: VGPR j, lanes 0-15 -> row j, lanes 16-31 -> row j+8.
#pragma unroll
  for (int mi = 0; mi < 4; ++mi)
#pragma unroll
    for (int j = 0; j < 8; ++j) {
      OutT* rowp = C + (size_t)(m0 + wm * 64 + mi * 16 + j + 8 * half) * N
                     + n0 + wn * 32 + lc;
      rowp[0]  = (OutT)acc[mi][0][j];
      rowp[16] = (OutT)acc[mi][1][j];
    }
}

// ---------------------------------------------------------------------------
// RoPE (in place on q_full / kv_full) + V transpose into v_t[b][h][dv][s].
// One block per (b,s,h), 64 threads. Thread j (<32) owns rope pair (j, j+32).
//   q_full: [b*S+s][h*192 + c]   (ld 3072)
//   kv_full:[b*S+s][h*320 + c]   (ld 5120)  (k:0..127, k_rope:128..191, v:192..319)
// ---------------------------------------------------------------------------
__global__ __launch_bounds__(64) void rope_prep(
    __bf16* __restrict__ qf, __bf16* __restrict__ kvf, __bf16* __restrict__ vt) {
  const int idx = blockIdx.x;                 // b*(S*H) + s*H + h
  const int h = idx & 15;
  const int s = (idx >> 4) & 2047;
  const int b = idx >> 15;
  const int t = threadIdx.x;
  const size_t row = (size_t)(b * 2048 + s);

  if (t < 32) {
    float inv = __powf(10000.0f, -(float)t / 32.0f);
    float ang = (float)s * inv;
    float c = __cosf(ang), sn = __sinf(ang);

    __bf16* qp = qf + row * 3072 + h * 192 + 128;
    float a0 = (float)qp[t], a1 = (float)qp[t + 32];
    qp[t]      = (__bf16)(a0 * c - a1 * sn);
    qp[t + 32] = (__bf16)(a1 * c + a0 * sn);

    __bf16* kp = kvf + row * 5120 + h * 320 + 128;
    float b0 = (float)kp[t], b1 = (float)kp[t + 32];
    kp[t]      = (__bf16)(b0 * c - b1 * sn);
    kp[t + 32] = (__bf16)(b1 * c + b0 * sn);
  }
  const __bf16* vsrc = kvf + row * 5120 + h * 320 + 192;
  __bf16* vdst = vt + (size_t)(b * 16 + h) * 128 * 2048 + s;
#pragma unroll
  for (int i = 0; i < 2; ++i) {
    int dv = t + 64 * i;
    vdst[(size_t)dv * 2048] = vsrc[dv];
  }
}

// ---------------------------------------------------------------------------
// Flash attention, causal. Grid (S/64, H, B), 128 threads = 4 waves.
// The causal block count (q0w+15)/64+1 == blockIdx.x+1 for ALL four waves,
// so the k-block loop is workgroup-uniform -> cooperative TDM staging:
// each wave DMAs its quarter of the K tile (16x192) and V tile (32x64) into
// ping-pong LDS with tensor_load_to_lds, waits its own TENSORcnt, barrier,
// then all WMMA fragments come from LDS while the next block's DMA flies.
// ---------------------------------------------------------------------------
__global__ __launch_bounds__(128) void mla_attn(
    const __bf16* __restrict__ qf, const __bf16* __restrict__ kvf,
    const __bf16* __restrict__ vt, __bf16* __restrict__ ctx) {
  const int b = blockIdx.z, h = blockIdx.y;
  const int wave = threadIdx.x >> 5, lane = threadIdx.x & 31;
  const int half = lane >> 4, lc = lane & 15;
  const int q0 = blockIdx.x * 64 + wave * 16;

  __shared__ __bf16 sK[2][64 * 192];         // K block, row-major (s x 192)
  __shared__ __bf16 sV[2][128 * 64];         // V block, row-major (dv x s)
  __shared__ __bf16 sP[4][16 * 64];          // per-wave P tile

  const __bf16* qb = qf + h * 192;                               // ld 3072
  const __bf16* kb = kvf + h * 320;                              // ld 5120
  const __bf16* vb = vt + (size_t)(b * 16 + h) * 128 * 2048;     // ld 2048
  const int qrow0 = b * 2048 + q0;

  auto issue_tiles = [&](int kbi, int buf) {
    const int k0 = kbi * 64;
    // K quarter: rows k0 + wave*16 .. +15, 192 contiguous bf16 each
    const __bf16* kg = kb + (size_t)(b * 2048 + k0 + wave * 16) * 5120;
    tdm_load_2d(&sK[buf][wave * 16 * 192], kg, 192, 16, 5120);
    // V quarter: dv rows wave*32 .. +31, 64 contiguous bf16 each at col k0
    const __bf16* vg = vb + (size_t)(wave * 32) * 2048 + k0;
    tdm_load_2d(&sV[buf][wave * 32 * 64], vg, 64, 32, 2048);
  };

  // Resident Q fragments: 16 rows x 192 = 6 chunks of K=32
  v16bf qfr[6];
#pragma unroll
  for (int kc = 0; kc < 6; ++kc) qfr[kc] = ldfrag(qb, qrow0, 3072, kc * 32);

  v8f o[8] = {};                 // 16 x 128 output accum
  float mrow[8], lrow[8];        // row stats for rows (j + 8*half)
#pragma unroll
  for (int j = 0; j < 8; ++j) { mrow[j] = -1e30f; lrow[j] = 0.0f; }

  const float scale = 0.07216878364870322f;   // 1/sqrt(192)
  const int nblk = blockIdx.x + 1;            // uniform causal bound

  issue_tiles(0, 0);
  for (int kbi = 0; kbi < nblk; ++kbi) {
    const int buf = kbi & 1;
    const int k0 = kbi * 64;

    __builtin_amdgcn_s_wait_tensorcnt(0);     // own DMAs done
    __syncthreads();                          // everyone's DMAs done
    if (kbi + 1 < nblk) issue_tiles(kbi + 1, buf ^ 1);

    // Scores: S = Q * K^T, 16x64, fp32 (K frags from LDS)
    v8f s[4];
#pragma unroll
    for (int nt = 0; nt < 4; ++nt) {
      v8f a = {};
#pragma unroll
      for (int kc = 0; kc < 6; ++kc) {
        v16bf kfr = ldfrag(sK[buf], nt * 16, 192, kc * 32);
        a = wmma_bf16(qfr[kc], kfr, a);
      }
      s[nt] = a;
    }

    // Causal mask + online softmax (row = q0 + j + 8*half, col = k0+nt*16+lc)
#pragma unroll
    for (int j = 0; j < 8; ++j) {
      const int mg = q0 + j + 8 * half;
      float t = -1e30f;
#pragma unroll
      for (int nt = 0; nt < 4; ++nt) {
        const int ng = k0 + nt * 16 + lc;
        float v = s[nt][j] * scale;
        v = (ng <= mg) ? v : -1e30f;
        s[nt][j] = v;
        t = fmaxf(t, v);
      }
#pragma unroll
      for (int off = 1; off < 16; off <<= 1)
        t = fmaxf(t, __shfl_xor(t, off, 32));
      const float mn  = fmaxf(mrow[j], t);
      const float scf = __expf(mrow[j] - mn);
      mrow[j] = mn;
      float rs = 0.0f;
#pragma unroll
      for (int nt = 0; nt < 4; ++nt) {
        float p = __expf(s[nt][j] - mn);
        s[nt][j] = p;
        rs += p;
      }
#pragma unroll
      for (int off = 1; off < 16; off <<= 1)
        rs += __shfl_xor(rs, off, 32);
      lrow[j] = lrow[j] * scf + rs;
#pragma unroll
      for (int dv = 0; dv < 8; ++dv) o[dv][j] *= scf;
    }

    // P -> LDS (bf16, 16x64 row-major); same-wave DS ops are in order.
    __bf16* pw = &sP[wave][0];
#pragma unroll
    for (int nt = 0; nt < 4; ++nt)
#pragma unroll
      for (int j = 0; j < 8; ++j)
        pw[(j + 8 * half) * 64 + nt * 16 + lc] = (__bf16)s[nt][j];

    v16bf pfr0 = ldfrag(pw, 0, 64, 0);
    v16bf pfr1 = ldfrag(pw, 0, 64, 32);

    // O += P * V   (V frags from LDS; rows = dv, K dim = sequence)
#pragma unroll
    for (int dv = 0; dv < 8; ++dv) {
      v16bf vf0 = ldfrag(sV[buf], dv * 16, 64, 0);
      o[dv] = wmma_bf16(pfr0, vf0, o[dv]);
      v16bf vf1 = ldfrag(sV[buf], dv * 16, 64, 32);
      o[dv] = wmma_bf16(pfr1, vf1, o[dv]);
    }
    __syncthreads();   // all reads of buf done before next iter's DMA reuses it
  }

  // Normalize and write context[b*S+s][h*128 + dv]
#pragma unroll
  for (int dv = 0; dv < 8; ++dv)
#pragma unroll
    for (int j = 0; j < 8; ++j) {
      const int r = b * 2048 + q0 + j + 8 * half;
      const int c = h * 128 + dv * 16 + lc;
      ctx[(size_t)r * 2048 + c] = (__bf16)(o[dv][j] / lrow[j]);
    }
}

// ---------------------------------------------------------------------------
// Launcher
// ---------------------------------------------------------------------------
extern "C" void kernel_launch(void* const* d_in, const int* in_sizes, int n_in,
                              void* d_out, int out_size, void* d_ws, size_t ws_size,
                              hipStream_t stream) {
  const float* x   = (const float*)d_in[0];
  // d_in[1] = mask (causality handled analytically)
  const float* wkd = (const float*)d_in[2];
  const float* wku = (const float*)d_in[3];
  const float* wqd = (const float*)d_in[4];
  const float* wqu = (const float*)d_in[5];
  const float* wo  = (const float*)d_in[6];

  char* ws = (char*)d_ws;
  size_t off = 0;
  auto alloc = [&](size_t elems) {
    __bf16* p = (__bf16*)(ws + off);
    off += ((elems * 2 + 255) & ~(size_t)255);
    return p;
  };
  __bf16* x_bf  = alloc((size_t)4096 * 2048);
  __bf16* wkdT  = alloc((size_t)2048 * 512);    // 512 x 2048 (N x K)
  __bf16* wkuT  = alloc((size_t)512 * 5120);    // 5120 x 512
  __bf16* wqdT  = alloc((size_t)2048 * 512);    // 512 x 2048
  __bf16* wquT  = alloc((size_t)512 * 3072);    // 3072 x 512
  __bf16* woT   = alloc((size_t)2048 * 2048);   // 2048 x 2048
  __bf16* c_kv  = alloc((size_t)4096 * 512);
  __bf16* c_q   = alloc((size_t)4096 * 512);
  __bf16* kvf   = alloc((size_t)4096 * 5120);   // (b*S+s) x (H*320)
  __bf16* qfull = alloc((size_t)4096 * 3072);   // (b*S+s) x (H*192)
  __bf16* vt    = alloc((size_t)2 * 16 * 128 * 2048);
  __bf16* ctx   = alloc((size_t)4096 * 2048);

  // x: plain convert; weights: transpose-convert (K x N -> N x K)
  f2bf<<<dim3(4096), 256, 0, stream>>>(x, x_bf, 4096 * 2048);
  f2bfT<<<dim3(512 / 32, 2048 / 32),  256, 0, stream>>>(wkd, wkdT, 2048, 512);
  f2bfT<<<dim3(5120 / 32, 512 / 32),  256, 0, stream>>>(wku, wkuT, 512, 5120);
  f2bfT<<<dim3(512 / 32, 2048 / 32),  256, 0, stream>>>(wqd, wqdT, 2048, 512);
  f2bfT<<<dim3(3072 / 32, 512 / 32),  256, 0, stream>>>(wqu, wquT, 512, 3072);
  f2bfT<<<dim3(2048 / 32, 2048 / 32), 256, 0, stream>>>(wo,  woT,  2048, 2048);

  // Down projections
  gemm_bf16_128x128<__bf16><<<dim3(4, 32),  256, 0, stream>>>(x_bf, wkdT, c_kv, 512, 2048);
  gemm_bf16_128x128<__bf16><<<dim3(4, 32),  256, 0, stream>>>(x_bf, wqdT, c_q,  512, 2048);
  // Up projections
  gemm_bf16_128x128<__bf16><<<dim3(40, 32), 256, 0, stream>>>(c_kv, wkuT, kvf,   5120, 512);
  gemm_bf16_128x128<__bf16><<<dim3(24, 32), 256, 0, stream>>>(c_q,  wquT, qfull, 3072, 512);
  // RoPE + V transpose
  rope_prep<<<dim3(65536), dim3(64), 0, stream>>>(qfull, kvf, vt);
  // Causal flash attention (TDM-staged K/V)
  mla_attn<<<dim3(32, 16, 2), dim3(128), 0, stream>>>(qfull, kvf, vt, ctx);
  // Output projection -> fp32 d_out
  gemm_bf16_128x128<float><<<dim3(16, 32), 256, 0, stream>>>(ctx, woT, (float*)d_out, 2048, 2048);
}
